// AllegroLayer_56109452755160
// MI455X (gfx1250) — matrix operand order repaired and checked by reading, not compiled, not run on hardware
//
#include <hip/hip_runtime.h>
#include <hip/hip_bf16.h>
#include <math.h>

// ---------------- WMMA types (CDNA5 / gfx1250, wave32) ----------------
typedef __attribute__((ext_vector_type(16))) __bf16 v16bf;
typedef __attribute__((ext_vector_type(8)))  float  v8f;
typedef __attribute__((ext_vector_type(4)))  unsigned int u32x4;
typedef __attribute__((ext_vector_type(4)))  int i32x4;
typedef __attribute__((ext_vector_type(8)))  int i32x8;

__device__ __forceinline__ v8f wmma_bf16(v16bf a, v16bf b, v8f c) {
  // D = A(16x32) * B(32x16) + C, f32 accumulate
  return __builtin_amdgcn_wmma_f32_16x16x32_bf16(false, a, false, b, (short)0, c, false, false);
}

// ---------------- problem constants ----------------
#define EDGES 32768
#define NODES 2048
#define DX    128
#define MUL   64
#define EPS_F 0.1f
#define S1_128 0.0883883476f   // 1/sqrt(128)
#define S1_320 0.0559016994f   // 1/sqrt(320)
#define S1_64  0.125f          // 1/sqrt(64)

// workspace byte offsets
#define WS_CG    0u
#define WS_PW    4096u              // packed bf16 weights base
#define W1P_OFF   0
#define W2AP_OFF  8192
#define W2BP_OFF  28672
#define W2CP_OFF  32768
#define WV1P_OFF  36864
#define WV2P_OFF  57344
#define WS_Y     163840u
#define WS_ENV   2260992u
#define WS_NODE  2392064u           // 2048*1024 floats = 8MB

// ---------------- CG tables: (l1,l2,l3, float-offset) ----------------
__device__ const int TRI[14][4] = {
  {1,1,2,0},{1,2,3,45},{0,0,0,150},{1,1,0,151},{2,2,0,160},
  {0,1,1,185},{1,0,1,194},{1,2,1,203},{2,1,1,248},{3,2,1,293},
  {0,2,2,398},{2,0,2,423},{2,2,2,448},{3,1,2,573}};
#define CG_TOTAL 678

// S path: pairs (0,0),(1,1),(2,2) -> l3=0
__device__ const int S_CGO[3] = {150,151,160};
__device__ const int S_D[3]   = {1,3,5};
__device__ const int S_OFF[3] = {0,1,4};
// V paths: path0 = l3=1 pairs (0,1),(1,0),(1,2),(2,1),(3,2)
//          path1 = l3=2 pairs (0,2),(1,1),(2,0),(2,2),(3,1)
__device__ const int VCGO[2][5] = {{185,194,203,248,293},{398,0,423,448,573}};
__device__ const int VD1 [2][5] = {{1,3,3,5,7},{1,3,5,5,7}};
__device__ const int VD2 [2][5] = {{3,1,5,3,5},{5,3,1,5,3}};
__device__ const int VOA [2][5] = {{0,1,1,4,9},{0,1,4,4,9}};
__device__ const int VOB [2][5] = {{1,0,4,1,4},{4,1,0,4,1}};

// ---------------- TDM: 1-D contiguous global -> LDS copy (n dwords, n<=65535) ----
// D# built per CDNA5 ISA: group0 {count=1, lds_addr, global_addr, type=2},
// group1 {data_size=4B, tensor_dim0=n, tensor_dim1=1, tile_dim0=n, tile_dim1=1, stride0=n}.
__device__ __forceinline__ void tdm_load_1d(const void* gsrc, void* lds_dst, unsigned n_dwords){
  unsigned long long ga = (unsigned long long)(size_t)gsrc;
  // low 32 bits of a flat LDS pointer are the LDS byte offset (aperture rule, ISA §10.2)
  unsigned lds = (unsigned)(size_t)lds_dst;
  unsigned n = n_dwords;
  u32x4 g0;
  g0[0] = 1u;                                              // count=1 (valid user D#)
  g0[1] = lds;                                             // lds_addr [63:32]
  g0[2] = (unsigned)(ga & 0xffffffffull);                  // global_addr[31:0]
  g0[3] = (unsigned)((ga >> 32) & 0x1ffffffull) | (2u<<30);// global_addr[56:32] | type=2
  i32x8 g1;
  g1[0] = (int)(2u << 16);                                 // data_size = 4 bytes
  g1[1] = (int)((n & 0xffffu) << 16);                      // tensor_dim0[15:0] @ bits[63:48]
  g1[2] = (int)((n >> 16) & 0xffffu) | (int)(1u << 16);    // tensor_dim0[31:16] | tensor_dim1=1
  g1[3] = (int)((n & 0xffffu) << 16);                      // tile_dim0 = n @ bits[127:112]
  g1[4] = 1;                                               // tile_dim1 = 1, tile_dim2 = 0
  g1[5] = (int)n;                                          // tensor_dim0_stride[31:0]
  g1[6] = 0;
  g1[7] = 0;
  i32x4 z4 = {0,0,0,0};
#if defined(__clang_major__) && (__clang_major__ >= 23)
  i32x8 z8 = {0,0,0,0,0,0,0,0};
  __builtin_amdgcn_tensor_load_to_lds(g0, g1, z4, z4, z8, 0);
#else
  __builtin_amdgcn_tensor_load_to_lds(g0, g1, z4, z4, 0);
#endif
}

// ---------------- on-device real Clebsch-Gordan build ----------------
struct cplx { double re, im; };
__device__ __forceinline__ cplx cmul(cplx a, cplx b){ return {a.re*b.re - a.im*b.im, a.re*b.im + a.im*b.re}; }
__device__ __forceinline__ cplx cconj(cplx a){ return {a.re, -a.im}; }
__device__ __forceinline__ double dfact(int n){ double r=1.0; for(int i=2;i<=n;i++) r*=(double)i; return r; }
__device__ __forceinline__ double dsign(int k){ return (k & 1) ? -1.0 : 1.0; }

__device__ double dev_w3j(int j1,int j2,int j3,int m1,int m2,int m3){
  if (m1+m2+m3 != 0) return 0.0;
  if (j3 < abs(j1-j2) || j3 > j1+j2) return 0.0;
  int t1=j2-m1-j3, t2=j1+m2-j3, t3=j1+j2-j3, t4=j1-m1, t5=j2+m2;
  int tmin = max(0, max(t1,t2)), tmax = min(t3, min(t4,t5));
  double s = 0.0;
  for (int t=tmin; t<=tmax; t++)
    s += dsign(t)/(dfact(t)*dfact(t-t1)*dfact(t-t2)*dfact(t3-t)*dfact(t4-t)*dfact(t5-t));
  double n = dsign(j1-j2-m3)
    * sqrt(dfact(j1+j2-j3)*dfact(j1-j2+j3)*dfact(-j1+j2+j3)/dfact(j1+j2+j3+1))
    * sqrt(dfact(j1+m1)*dfact(j1-m1)*dfact(j2+m2)*dfact(j2-m2)*dfact(j3+m3)*dfact(j3-m3));
  return n*s;
}

// row a of complex->real basis change U(l): at most 2 nonzeros
__device__ int urow(int l, int a, int* col, cplx* val){
  const double s2 = 0.70710678118654752440;
  int m = a - l;
  if (m == 0){ col[0]=l; val[0]={1.0,0.0}; return 1; }
  if (m > 0){ col[0]=l+m; val[0]={dsign(m)*s2,0.0}; col[1]=l-m; val[1]={s2,0.0}; return 2; }
  m = -m;
  col[0]=l-m; val[0]={0.0,s2}; col[1]=l+m; val[1]={0.0,-dsign(m)*s2}; return 2;
}

__global__ void cg_init_kernel(float* __restrict__ cg_out){
  int t = threadIdx.x;
  if (t >= 14) return;
  int l1=TRI[t][0], l2=TRI[t][1], l3=TRI[t][2], off=TRI[t][3];
  int d1=2*l1+1, d2=2*l2+1, d3=2*l3+1;
  double Cre[125], Cim[125];
  double sre=0.0, sim=0.0;
  for(int a=0;a<d1;a++){
    int ca[2]; cplx va[2]; int na=urow(l1,a,ca,va);
    for(int b=0;b<d2;b++){
      int cb[2]; cplx vb[2]; int nb=urow(l2,b,cb,vb);
      for(int c=0;c<d3;c++){
        int cc2[2]; cplx vc[2]; int nc=urow(l3,c,cc2,vc);
        double are=0.0, aim=0.0;
        for(int ia=0;ia<na;ia++) for(int ib=0;ib<nb;ib++) for(int ic=0;ic<nc;ic++){
          int m1=ca[ia]-l1, m2=cb[ib]-l2, m3=cc2[ic]-l3;
          if (m1+m2 != m3) continue;
          double ccv = dsign(l1-l2+m3)*sqrt((double)d3)*dev_w3j(l1,l2,l3,m1,m2,-m3);
          if (ccv == 0.0) continue;
          cplx uu = cmul(cmul(va[ia], vb[ib]), cconj(vc[ic]));
          are += uu.re*ccv; aim += uu.im*ccv;
        }
        int idx=(a*d2+b)*d3+c;
        Cre[idx]=are; Cim[idx]=aim; sre+=fabs(are); sim+=fabs(aim);
      }
    }
  }
  bool useim = (sim > sre);
  int tot = d1*d2*d3;
  double nrm = 0.0;
  for(int i=0;i<tot;i++){ double v = useim?Cim[i]:Cre[i]; nrm += v*v; }
  double scl = sqrt((double)d3)/sqrt(nrm);
  for(int i=0;i<tot;i++){ double v = useim?Cim[i]:Cre[i]; cg_out[off+i] = (float)(v*scl); }
}

// ---------------- weight packing into B-fragment order ----------------
// B-operand layout (ISA 7.12.2): element idx of lane L -> k = kc*32 + (idx<8 ? 8*(L>>4)+idx : 16+8*(L>>4)+idx-8), n = nt*16 + (L&15)
__global__ __launch_bounds__(256) void pack_weights_kernel(const float* __restrict__ W,
                                                           __bf16* __restrict__ dst, int total){
  int i = blockIdx.x*256 + threadIdx.x;
  if (i >= total) return;
  int idx = i & 15, lane = (i>>4) & 31, nt = (i>>9) & 3, kc = i>>11;
  int h = lane>>4, n = nt*16 + (lane&15);
  int k = kc*32 + ((idx<8) ? (8*h+idx) : (16+8*h+(idx-8)));
  dst[i] = (__bf16)W[(size_t)k*64 + n];
}

// ---------------- spherical harmonics + envelope ----------------
__global__ __launch_bounds__(256) void ylm_kernel(const float* __restrict__ vectors,
                                                  const float* __restrict__ cg,
                                                  float* __restrict__ Y,
                                                  float* __restrict__ env){
  int e = blockIdx.x*256 + threadIdx.x;   // grid covers EDGES exactly
  float vx=vectors[3*e+0], vy=vectors[3*e+1], vz=vectors[3*e+2];
  float d = sqrtf(vx*vx+vy*vy+vz*vz);
  float inv = 1.f/d;
  float y1[3] = {1.73205081f*vy*inv, 1.73205081f*vz*inv, 1.73205081f*vx*inv};
  float t2[5] = {0,0,0,0,0};
  for(int a=0;a<3;a++) for(int b=0;b<3;b++){
    float p = y1[a]*y1[b];
    for(int c=0;c<5;c++) t2[c] += cg[(a*3+b)*5+c]*p;
  }
  float n2=0.f; for(int c=0;c<5;c++) n2+=t2[c]*t2[c];
  float s2 = 2.23606798f*rsqrtf(n2);
  float y2[5]; for(int c=0;c<5;c++) y2[c]=t2[c]*s2;
  const float* cg123 = cg + 45;
  float t3[7] = {0,0,0,0,0,0,0};
  for(int a=0;a<3;a++) for(int b=0;b<5;b++){
    float p = y1[a]*y2[b];
    for(int c=0;c<7;c++) t3[c] += cg123[(a*5+b)*7+c]*p;
  }
  float n3=0.f; for(int c=0;c<7;c++) n3+=t3[c]*t3[c];
  float s3 = 2.64575131f*rsqrtf(n3);
  float* yr = Y + (size_t)e*16;
  yr[0]=1.f;
  for(int c=0;c<3;c++) yr[1+c]=y1[c];
  for(int c=0;c<5;c++) yr[4+c]=y2[c];
  for(int c=0;c<7;c++) yr[9+c]=t3[c]*s3;
  // polynomial envelope p=6: 1 - 28 d^6 + 48 d^7 - 21 d^8
  float d2=d*d, d6=d2*d2*d2;
  float f = 1.f - 28.f*d6 + 48.f*d6*d - 21.f*d6*d2;
  env[e] = (d < 1.f) ? f : 0.f;
}

// ---------------- w = x@W1/sqrt(128) via WMMA, then scatter w*Y into node ----------------
__global__ __launch_bounds__(256) void edge_w_scatter_kernel(const float* __restrict__ x,
                                                             const __bf16* __restrict__ W1p,
                                                             const int* __restrict__ senders,
                                                             const float* __restrict__ Y,
                                                             float* __restrict__ node){
  const int wv = threadIdx.x>>5, lane = threadIdx.x & 31;
  const int eBase = (blockIdx.x*8 + wv)*16;
  const int m = lane & 15, h = lane>>4;
  v8f acc[4] = {};
  for(int kc=0;kc<4;kc++){
    // A-operand: rows of x; layout k = kc*32 + 8h + idx(0..7), and +16 for idx 8..15 -> contiguous float4s
    const float* xr = x + (size_t)(eBase+m)*DX + kc*32 + 8*h;
    float4 p0 = *(const float4*)(xr);
    float4 p1 = *(const float4*)(xr+4);
    float4 q0 = *(const float4*)(xr+16);
    float4 q1 = *(const float4*)(xr+20);
    v16bf a;
    a[0]=(__bf16)p0.x; a[1]=(__bf16)p0.y; a[2]=(__bf16)p0.z; a[3]=(__bf16)p0.w;
    a[4]=(__bf16)p1.x; a[5]=(__bf16)p1.y; a[6]=(__bf16)p1.z; a[7]=(__bf16)p1.w;
    a[8]=(__bf16)q0.x; a[9]=(__bf16)q0.y; a[10]=(__bf16)q0.z; a[11]=(__bf16)q0.w;
    a[12]=(__bf16)q1.x; a[13]=(__bf16)q1.y; a[14]=(__bf16)q1.z; a[15]=(__bf16)q1.w;
    #pragma unroll
    for(int nt=0;nt<4;nt++){
      v16bf b = *(const v16bf*)(W1p + (size_t)((kc*4+nt)*32+lane)*16);
      acc[nt] = wmma_bf16(a, b, acc[nt]);
    }
  }
  const int col = lane & 15;
  #pragma unroll
  for(int r=0;r<8;r++){
    int e = eBase + r + 8*h;          // D layout: M = r + 8*(lane>>4)
    int s = senders[e];
    const float* yrp = Y + (size_t)e*16;
    float yv[16];
    #pragma unroll
    for(int l=0;l<16;l++) yv[l] = yrp[l];
    float* nb = node + (size_t)s*1024;
    #pragma unroll
    for(int nt=0;nt<4;nt++){
      float wval = acc[nt][r]*S1_128;
      float* dst = nb + (nt*16+col)*16;
      for(int l=0;l<16;l++) atomicAdd(dst+l, wval*yv[l]);
    }
  }
}

// ---------------- fused main kernel ----------------
// A-operand fragment built from an LDS f32 row (ISA mapping)
__device__ __forceinline__ v16bf afrag_f32(const float* rb, int kc, int h){
  v16bf a;
  #pragma unroll
  for(int idx=0;idx<16;idx++){
    int k = kc*32 + ((idx<8) ? (8*h+idx) : (16+8*h+(idx-8)));
    a[idx] = (__bf16)rb[k];
  }
  return a;
}

__device__ __forceinline__ void mlp_layer(const float* in, int stride, int KC,
                                          const __bf16* Wp, float scale, bool act,
                                          float* outLds, int nt, int lane){
  v8f acc = {};
  const int h = lane>>4, m = lane & 15, col = lane & 15;
  const float* rb = in + m*stride;
  for(int kc=0;kc<KC;kc++){
    v16bf a = afrag_f32(rb, kc, h);
    v16bf b = *(const v16bf*)(Wp + (size_t)((kc*4+nt)*32+lane)*16);
    acc = wmma_bf16(a, b, acc);
  }
  #pragma unroll
  for(int r=0;r<8;r++){
    float vv = acc[r]*scale;
    if (act) vv = vv*(1.f/(1.f+__expf(-vv)));   // silu
    outLds[(r+8*h)*64 + nt*16 + col] = vv;
  }
}

// shared-mem byte offsets (dynamic LDS, 133920 B total -> 2 blocks/WGP on 320KB)
#define SH_CG    0u
#define SH_SND   2720u
#define SH_ENV   2784u
#define SH_A     2848u      // 16*1024 bf16 = 32768
#define SH_V     35616u     // 16*576  f32  = 36864
#define SH_X2    72480u     // 16*320  f32  = 20480
#define SH_H     92960u     // 16*64   f32  = 4096
#define SH_H2    97056u     // 16*64   f32  = 4096
#define SH_VST   101152u    // 16*512  f32  = 32768
#define SH_TOTAL 133920u

__global__ __launch_bounds__(256) void allegro_main_kernel(
    const float* __restrict__ x, const float* __restrict__ V,
    const int* __restrict__ senders, const float* __restrict__ node,
    const float* __restrict__ cg_g, const float* __restrict__ envArr,
    const __bf16* __restrict__ W2a_p, const __bf16* __restrict__ W2b_p,
    const __bf16* __restrict__ W2c_p, const __bf16* __restrict__ Wv1_p,
    const __bf16* __restrict__ Wv2_p, float* __restrict__ out){
  extern __shared__ char smem[];
  float*  cgS    = (float*)(smem + SH_CG);
  int*    snd    = (int*)  (smem + SH_SND);
  float*  envsh  = (float*)(smem + SH_ENV);
  __bf16* Ash    = (__bf16*)(smem + SH_A);
  float*  Vsh    = (float*)(smem + SH_V);
  float*  x2sh   = (float*)(smem + SH_X2);
  float*  hsh    = (float*)(smem + SH_H);
  float*  h2sh   = (float*)(smem + SH_H2);
  float*  Vstage = (float*)(smem + SH_VST);

  const int tid = threadIdx.x;
  const int wv = tid>>5, lane = tid & 31;
  const int eBase = blockIdx.x*16;

  // -------- phase 0: TDM the V tile (wave 0), cooperatively stage the rest --------
  if (wv == 0) {
    // 16 consecutive edge rows of V are one contiguous 36864B slab -> single TDM descriptor
    tdm_load_1d(V + (size_t)eBase*576, Vsh, 16*576);
  }
  for(int i=tid;i<CG_TOTAL;i+=256) cgS[i] = cg_g[i];
  if (tid < 16){ snd[tid] = senders[eBase+tid]; envsh[tid] = envArr[eBase+tid]; }
  __syncthreads();
  for(int i=tid;i<16*1024;i+=256){
    int e=i>>10, j=i&1023;
    Ash[i] = (__bf16)(node[(size_t)snd[e]*1024 + j]*EPS_F);
  }
  for(int i=tid;i<16*128;i+=256){
    int e=i>>7, j=i&127;
    x2sh[e*320 + j] = x[(size_t)(eBase+e)*DX + j];
  }
  if (wv == 0) __builtin_amdgcn_s_wait_tensorcnt(0);   // V tile resident in LDS
  __syncthreads();

  // -------- phase 1: S scalars -> x2 columns 128..319 --------
  for(int i=tid;i<16*192;i+=256){
    int e=i/192, u=i-192*e, chan=u/3, p=u-3*chan;
    const float* cg = cgS + S_CGO[p];
    const __bf16* Ar = Ash + e*1024 + chan*16 + S_OFF[p];
    const float*  Br = Vsh + e*576  + chan*9  + S_OFF[p];
    int d = S_D[p];
    float acc=0.f;
    for(int a=0;a<d;a++){
      float av=(float)Ar[a];
      for(int b=0;b<d;b++) acc += cg[a*d+b]*av*Br[b];
    }
    x2sh[e*320 + 128 + u] = acc;
  }
  __syncthreads();

  // -------- phase 2: MLP (waves 0..3 each own one 16-col n-tile) --------
  if (wv < 4) mlp_layer(x2sh, 320, 10, W2a_p, S1_320, true, hsh, wv, lane);
  __syncthreads();
  if (wv < 4) mlp_layer(hsh, 64, 2, W2b_p, S1_64, true, h2sh, wv, lane);
  __syncthreads();
  if (wv < 4){
    // layer 3: no act, * envelope, store x_out (non-temporal: written once)
    v8f acc = {};
    const int h = lane>>4, m = lane & 15, col = lane & 15;
    const float* rb = h2sh + m*64;
    for(int kc=0;kc<2;kc++){
      v16bf a = afrag_f32(rb, kc, h);
      v16bf b = *(const v16bf*)(W2c_p + (size_t)((kc*4+wv)*32+lane)*16);
      acc = wmma_bf16(a, b, acc);
    }
    #pragma unroll
    for(int r=0;r<8;r++){
      int mm = r + 8*h;
      __builtin_nontemporal_store(acc[r]*S1_64*envsh[mm],
                                  &out[(size_t)(eBase+mm)*64 + wv*16 + col]);
    }
  }

  // -------- phase 3: V1/V2 tensor-product GEMMs (wave -> one component c) --------
  {
    const int path = (wv < 3) ? 0 : 1;          // 0: l3=1 (3 comps), 1: l3=2 (5 comps)
    const int c    = path ? (wv-3) : wv;
    const int d3   = path ? 5 : 3;
    const __bf16* Wp = path ? Wv2_p : Wv1_p;
    const int h = lane>>4, m = lane & 15;
    const __bf16* Arow0 = Ash + m*1024;
    const float*  Vrow0 = Vsh + m*576;
    v8f acc[4] = {};
    for(int kc=0;kc<10;kc++){
      v16bf a;
      for(int idx=0;idx<16;idx++){
        int k = kc*32 + ((idx<8) ? (8*h+idx) : (16+8*h+(idx-8)));  // u index in 0..319
        int chan = k/5, p = k-5*chan;                               // u = chan*5 + pair
        const float* cg = cgS + VCGO[path][p] + c;
        const int d1 = VD1[path][p], d2 = VD2[path][p];
        const __bf16* Ar = Arow0 + chan*16 + VOA[path][p];
        const float*  Br = Vrow0 + chan*9  + VOB[path][p];
        float tp = 0.f;
        for(int aa=0;aa<d1;aa++){
          float av = (float)Ar[aa];
          const float* cr = cg + aa*d2*d3;
          for(int bb=0;bb<d2;bb++) tp += cr[bb*d3]*av*Br[bb];
        }
        a[idx] = (__bf16)tp;
      }
      #pragma unroll
      for(int nt=0;nt<4;nt++){
        v16bf b = *(const v16bf*)(Wp + (size_t)((kc*4+nt)*32+lane)*16);
        acc[nt] = wmma_bf16(a, b, acc[nt]);
      }
    }
    #pragma unroll
    for(int nt=0;nt<4;nt++){
      #pragma unroll
      for(int r=0;r<8;r++){
        int mm = r + 8*h, n = nt*16 + (lane & 15);
        int j = path ? (192 + n*5 + c) : (n*3 + c);   // V_out layout: [V1o (64*3) | V2e (64*5)]
        Vstage[mm*512 + j] = acc[nt][r]*S1_320;
      }
    }
  }
  __syncthreads();

  // -------- phase 4: coalesced non-temporal V_out store --------
  const size_t vbase = (size_t)EDGES*64;
  for(int i=tid;i<16*512;i+=256){
    int e=i>>9, j=i&511;
    __builtin_nontemporal_store(Vstage[i], &out[vbase + (size_t)(eBase+e)*512 + j]);
  }
}

// ---------------- launcher ----------------
extern "C" void kernel_launch(void* const* d_in, const int* in_sizes, int n_in,
                              void* d_out, int out_size, void* d_ws, size_t ws_size,
                              hipStream_t stream) {
  (void)in_sizes; (void)n_in; (void)out_size; (void)ws_size;
  const float* vectors = (const float*)d_in[0];
  const float* x       = (const float*)d_in[1];
  const float* V       = (const float*)d_in[2];
  const int*   senders = (const int*)  d_in[3];
  // d_in[4] = species (unused by the layer body)
  const float* W1  = (const float*)d_in[5];
  const float* W2a = (const float*)d_in[6];
  const float* W2b = (const float*)d_in[7];
  const float* W2c = (const float*)d_in[8];
  const float* Wv1 = (const float*)d_in[9];
  const float* Wv2 = (const float*)d_in[10];
  float* out = (float*)d_out;

  char* ws = (char*)d_ws;
  float*  cg_d = (float*)(ws + WS_CG);
  __bf16* pw   = (__bf16*)(ws + WS_PW);
  float*  Yd   = (float*)(ws + WS_Y);
  float*  envd = (float*)(ws + WS_ENV);
  float*  node = (float*)(ws + WS_NODE);

  // 1) build real-CG tables on device (capture-safe, deterministic)
  cg_init_kernel<<<1, 32, 0, stream>>>(cg_d);
  // 2) pack all weights into WMMA B-fragment order (bf16)
  pack_weights_kernel<<<(128*64+255)/256, 256, 0, stream>>>(W1,  pw+W1P_OFF,  128*64);
  pack_weights_kernel<<<(320*64+255)/256, 256, 0, stream>>>(W2a, pw+W2AP_OFF, 320*64);
  pack_weights_kernel<<<( 64*64+255)/256, 256, 0, stream>>>(W2b, pw+W2BP_OFF,  64*64);
  pack_weights_kernel<<<( 64*64+255)/256, 256, 0, stream>>>(W2c, pw+W2CP_OFF,  64*64);
  pack_weights_kernel<<<(320*64+255)/256, 256, 0, stream>>>(Wv1, pw+WV1P_OFF, 320*64);
  pack_weights_kernel<<<(320*64+255)/256, 256, 0, stream>>>(Wv2, pw+WV2P_OFF, 320*64);
  // 3) zero the L2-resident node accumulator (8MB)
  hipMemsetAsync(node, 0, (size_t)NODES*1024*sizeof(float), stream);
  // 4) spherical harmonics + envelope per edge
  ylm_kernel<<<EDGES/256, 256, 0, stream>>>(vectors, cg_d, Yd, envd);
  // 5) w = x@W1 (WMMA) + atomic scatter of w*Y into node
  edge_w_scatter_kernel<<<EDGES/128, 256, 0, stream>>>(x, pw+W1P_OFF, senders, Yd, node);
  // 6) fused gather + tensor products + MLP + V-channel GEMMs (TDM stages V tile)
  allegro_main_kernel<<<EDGES/16, 256, SH_TOTAL, stream>>>(
      x, V, senders, node, cg_d, envd,
      pw+W2AP_OFF, pw+W2BP_OFF, pw+W2CP_OFF, pw+WV1P_OFF, pw+WV2P_OFF, out);
}
